// MixtureOfExperts_50835232916163
// MI455X (gfx1250) — compile-verified, matrix operand order checked by reading, not run on hardware
//
#include <hip/hip_runtime.h>

// ---------------------------------------------------------------------------
// Dense soft-MoE for MI455X (gfx1250, wave32, WMMA).
// B=8192, D=1024, H=1024, E=8, G=256. All big GEMMs run as bf16 WMMA
// (v_wmma_f32_16x16x32_bf16) with f32 accumulation.
// Tiles: 128x128 per block, K staged 64-deep into double-buffered LDS via
// GLOBAL_LOAD_ASYNC_TO_LDS_B128 (ASYNCcnt), so next-stage fetch overlaps the
// current stage's 16 WMMAs.
// ---------------------------------------------------------------------------

enum { Bc = 8192, Dc = 1024, Hc = 1024, Ec = 8, Gc = 256 };

typedef __attribute__((ext_vector_type(16))) __bf16 v16bf;
typedef __attribute__((ext_vector_type(8)))  float  v8f;

// ---- bf16 helpers ---------------------------------------------------------
__device__ __forceinline__ unsigned short f2bf(float f) {
  unsigned u = __float_as_uint(f);
  u += 0x7fffu + ((u >> 16) & 1u);          // round to nearest even
  return (unsigned short)(u >> 16);
}
__device__ __forceinline__ float bf2f(unsigned short u) {
  return __uint_as_float(((unsigned)u) << 16);
}

// ---- convert kernels ------------------------------------------------------
__global__ __launch_bounds__(256) void k_f2bf4(const float* __restrict__ s,
                                               unsigned short* __restrict__ d) {
  long i = (long)blockIdx.x * 256 + threadIdx.x;     // index of float4
  float4 v = reinterpret_cast<const float4*>(s)[i];
  ushort4 o;
  o.x = f2bf(v.x); o.y = f2bf(v.y); o.z = f2bf(v.z); o.w = f2bf(v.w);
  reinterpret_cast<ushort4*>(d)[i] = o;
}

// W2 [E,D,H] (f32) -> W2t [D, E*H] (bf16): W2t[d][e*H+h] = W2[e][d][h]
__global__ __launch_bounds__(256) void k_w2t(const float* __restrict__ W2,
                                             unsigned short* __restrict__ W2t) {
  long i = (long)blockIdx.x * 256 + threadIdx.x;     // over E*D*H
  int  e  = (int)(i >> 20);                          // D*H = 2^20
  long r  = i & ((1l << 20) - 1);
  int  dd = (int)(r >> 10);
  int  hh = (int)(r & 1023);
  W2t[((long)dd << 13) + (e << 10) + hh] = f2bf(W2[i]);
}

// ---- gating: g = softmax(relu(x Wg1^T + bg1) Wg2^T + bg2) -----------------
__global__ __launch_bounds__(256) void k_gate(const float* __restrict__ x,
                                              const float* __restrict__ Wg1,
                                              const float* __restrict__ bg1,
                                              const float* __restrict__ Wg2,
                                              const float* __restrict__ bg2,
                                              float* __restrict__ gate) {
  __shared__ __align__(16) float xs[Dc];
  __shared__ float gh[Gc];
  __shared__ float lg[Ec];
  const int b = blockIdx.x, t = threadIdx.x;
  reinterpret_cast<float4*>(xs)[t] =
      reinterpret_cast<const float4*>(x + (long)b * Dc)[t];
  __syncthreads();
  float acc = bg1[t];
  const float4* wr  = reinterpret_cast<const float4*>(Wg1 + (long)t * Dc);
  const float4* xs4 = reinterpret_cast<const float4*>(xs);
  for (int k = 0; k < Dc / 4; ++k) {
    float4 a = xs4[k], w = wr[k];
    acc += a.x * w.x + a.y * w.y + a.z * w.z + a.w * w.w;
  }
  gh[t] = acc > 0.f ? acc : 0.f;
  __syncthreads();
  if (t < Ec) {
    float l = bg2[t];
    for (int g = 0; g < Gc; ++g) l += gh[g] * Wg2[t * Gc + g];
    lg[t] = l;
  }
  __syncthreads();
  if (t < Ec) {
    float m = lg[0];
    for (int i = 1; i < Ec; ++i) m = fmaxf(m, lg[i]);
    float s = 0.f;
    for (int i = 0; i < Ec; ++i) s += expf(lg[i] - m);
    gate[(long)b * Ec + t] = expf(lg[t] - m) / s;
  }
}

// gb[b,d] = sum_e gate[b,e] * b2[e,d]
__global__ __launch_bounds__(256) void k_gbias(const float* __restrict__ gate,
                                               const float* __restrict__ b2,
                                               float* __restrict__ gb) {
  long i = (long)blockIdx.x * 256 + threadIdx.x;     // over B*D
  int  b = (int)(i >> 10), d = (int)(i & 1023);
  float s = 0.f;
  for (int e = 0; e < Ec; ++e) s += gate[(long)b * Ec + e] * b2[e * Dc + d];
  gb[i] = s;
}

// ---- WMMA GEMM: C[M,N] = A[M,K] * B[N,K]^T (both K-contiguous bf16) -------
#define BM 128
#define BN 128
#define BKE 64     // K elements per LDS stage (2 WMMA sub-steps)
#define LSTR 144   // LDS row stride bytes (64 bf16 = 128B data + 16B pad)

struct B32x8 { uint4 lo, hi; };

__device__ __forceinline__ v16bf ldsFrag(const unsigned char* lds, int off) {
  B32x8 t;
  t.lo = *reinterpret_cast<const uint4*>(lds + off);
  t.hi = *reinterpret_cast<const uint4*>(lds + off + 32);
  return __builtin_bit_cast(v16bf, t);
}

// 64B per lane: 4x async b128, instruction offset applies to BOTH LDS+global.
__device__ __forceinline__ void asyncLoad64B(unsigned ldsOff,
                                             const unsigned short* g) {
  unsigned long long ga = (unsigned long long)(uintptr_t)g;
  asm volatile(
      "global_load_async_to_lds_b128 %0, %1, off\n\t"
      "global_load_async_to_lds_b128 %0, %1, off offset:16\n\t"
      "global_load_async_to_lds_b128 %0, %1, off offset:32\n\t"
      "global_load_async_to_lds_b128 %0, %1, off offset:48"
      :: "v"(ldsOff), "v"(ga) : "memory");
}

// EPI 0: out=bf16 relu(acc + bias[n]) ; EPI 1: out=bf16 (acc + bias[n])
// EPI 2: out=f32 (acc + bias[m*N+n])  (bias = precomputed gate*b2 map)
template <int EPI>
__global__ __launch_bounds__(256) void moe_gemm(
    const unsigned short* __restrict__ A, const unsigned short* __restrict__ Bw,
    const float* __restrict__ bias, unsigned short* __restrict__ outB,
    float* __restrict__ outF, int M, int N, int K,
    long aStep, long bStep, long biasStep, long outStep) {
  __shared__ __align__(16) unsigned char stA[2][BM * LSTR];
  __shared__ __align__(16) unsigned char stB[2][BN * LSTR];

  const int e = blockIdx.z;
  A    += (long)e * aStep;
  Bw   += (long)e * bStep;
  bias += (long)e * biasStep;
  if (EPI != 2) outB += (long)e * outStep;

  const int tid  = threadIdx.x;
  const int lane = tid & 31;
  const int wave = tid >> 5;
  const int wm   = wave & 3;   // M offset wm*32
  const int wn   = wave >> 2;  // N offset wn*64

  const long m0 = (long)blockIdx.y * BM;
  const long n0 = (long)blockIdx.x * BN;

  // cooperative stage loaders: 2 threads per row, 32 bf16 (64B) each
  const int lr = tid >> 1;
  const int lh = tid & 1;
  const unsigned short* aSrc = A  + (m0 + lr) * (long)K + lh * 32;
  const unsigned short* bSrc = Bw + (n0 + lr) * (long)K + lh * 32;
  unsigned aOffL[2], bOffL[2];
#pragma unroll
  for (int s = 0; s < 2; ++s) {
    aOffL[s] = (unsigned)(uintptr_t)(&stA[s][lr * LSTR + lh * 64]);
    bOffL[s] = (unsigned)(uintptr_t)(&stB[s][lr * LSTR + lh * 64]);
  }

  v8f acc[2][4];
#pragma unroll
  for (int i = 0; i < 2; ++i)
#pragma unroll
    for (int j = 0; j < 4; ++j) {
      v8f z;
#pragma unroll
      for (int v = 0; v < 8; ++v) z[v] = 0.f;
      acc[i][j] = z;
    }

  const int halfOff = (lane >> 4) * 16;  // byte offset selecting K-half
  const int rA      = lane & 15;
  const int T       = K / BKE;

  // prologue: stage 0 in flight
  asyncLoad64B(aOffL[0], aSrc);
  asyncLoad64B(bOffL[0], bSrc);

  for (int it = 0; it < T; ++it) {
    const int cur = it & 1;
    if (it + 1 < T) {
      const int nxt = cur ^ 1;
      asyncLoad64B(aOffL[nxt], aSrc + (long)(it + 1) * BKE);
      asyncLoad64B(bOffL[nxt], bSrc + (long)(it + 1) * BKE);
      // in-order completion: <=8 outstanding => current stage resident
      asm volatile("s_wait_asynccnt 0x8" ::: "memory");
    } else {
      asm volatile("s_wait_asynccnt 0x0" ::: "memory");
    }
    __syncthreads();

    const unsigned char* As = stA[cur];
    const unsigned char* Bs = stB[cur];
#pragma unroll
    for (int sub = 0; sub < 2; ++sub) {
      v16bf af[2], bfr[4];
#pragma unroll
      for (int i = 0; i < 2; ++i)
        af[i] = ldsFrag(As, (wm * 32 + i * 16 + rA) * LSTR + sub * 64 + halfOff);
#pragma unroll
      for (int j = 0; j < 4; ++j)
        bfr[j] = ldsFrag(Bs, (wn * 64 + j * 16 + rA) * LSTR + sub * 64 + halfOff);
#pragma unroll
      for (int i = 0; i < 2; ++i)
#pragma unroll
        for (int j = 0; j < 4; ++j)
          acc[i][j] = __builtin_amdgcn_wmma_f32_16x16x32_bf16(
              false, af[i], false, bfr[j], (short)0, acc[i][j], false, false);
    }
    __syncthreads();   // all reads done before next-stage overwrite
  }

  // epilogue: C layout = lane(L%16)=N, VGPR v + 8*(L/16) = M
  const int mB = (lane >> 4) * 8;
  const int nI = lane & 15;
#pragma unroll
  for (int i = 0; i < 2; ++i)
#pragma unroll
    for (int j = 0; j < 4; ++j) {
      long mg0 = m0 + wm * 32 + i * 16 + mB;
      long ng  = n0 + wn * 64 + j * 16 + nI;
#pragma unroll
      for (int v = 0; v < 8; ++v) {
        long  mg  = mg0 + v;
        float val = acc[i][j][v];
        if (EPI == 0) {
          val += bias[ng];
          val = val > 0.f ? val : 0.f;
          outB[mg * (long)N + ng] = f2bf(val);
        } else if (EPI == 1) {
          val += bias[ng];
          outB[mg * (long)N + ng] = f2bf(val);
        } else {
          val += bias[mg * (long)N + ng];
          outF[mg * (long)N + ng] = val;
        }
      }
    }
}

// ---- LayerNorm + ReLU + residual + gate-scale -> bf16 h3[b, e*H+f] --------
__global__ __launch_bounds__(256) void k_ln(const unsigned short* __restrict__ z,
                                            const unsigned short* __restrict__ h,
                                            const float* __restrict__ ln_g,
                                            const float* __restrict__ ln_b,
                                            const float* __restrict__ gate,
                                            unsigned short* __restrict__ h3) {
  const int b = blockIdx.x, e = blockIdx.y, t = threadIdx.x;
  __shared__ float r1[256], r2[256];
  const long row = ((long)e * Bc + b) * Hc;
  float zv[4], hv[4], s = 0.f, s2 = 0.f;
#pragma unroll
  for (int i = 0; i < 4; ++i) {
    int f = t * 4 + i;
    float zz = bf2f(z[row + f]);
    zv[i] = zz; s += zz; s2 += zz * zz;
    hv[i] = bf2f(h[row + f]);
  }
  r1[t] = s; r2[t] = s2;
  __syncthreads();
  for (int st = 128; st > 0; st >>= 1) {
    if (t < st) { r1[t] += r1[t + st]; r2[t] += r2[t + st]; }
    __syncthreads();
  }
  const float mu  = r1[0] * (1.f / Hc);
  const float var = r2[0] * (1.f / Hc) - mu * mu;
  const float inv = rsqrtf(var + 1e-5f);
  const float gv  = gate[(long)b * Ec + e];
  const long  orow = ((long)b * Ec + e) * Hc;
#pragma unroll
  for (int i = 0; i < 4; ++i) {
    int f = t * 4 + i;
    float zn = (zv[i] - mu) * inv * ln_g[e * Hc + f] + ln_b[e * Hc + f];
    float r  = (zn > 0.f ? zn : 0.f) + hv[i];
    h3[orow + f] = f2bf(r * gv);
  }
}

// ---------------------------------------------------------------------------
extern "C" void kernel_launch(void* const* d_in, const int* in_sizes, int n_in,
                              void* d_out, int out_size, void* d_ws, size_t ws_size,
                              hipStream_t stream) {
  (void)in_sizes; (void)n_in; (void)out_size; (void)ws_size;
  const float* x   = (const float*)d_in[0];
  const float* W0  = (const float*)d_in[1];
  const float* b0  = (const float*)d_in[2];
  const float* W1  = (const float*)d_in[3];
  const float* b1  = (const float*)d_in[4];
  const float* lng = (const float*)d_in[5];
  const float* lnb = (const float*)d_in[6];
  const float* W2  = (const float*)d_in[7];
  const float* b2  = (const float*)d_in[8];
  const float* Wg1 = (const float*)d_in[9];
  const float* bg1 = (const float*)d_in[10];
  const float* Wg2 = (const float*)d_in[11];
  const float* bg2 = (const float*)d_in[12];
  float* out = (float*)d_out;

  // workspace carve-up (~480 MB)
  char* w = (char*)d_ws;
  auto carve = [&](size_t bytes) {
    char* p = w; w += (bytes + 255) & ~(size_t)255; return p;
  };
  unsigned short* xb  = (unsigned short*)carve((size_t)Bc * Dc * 2);        // 16MB
  unsigned short* W0b = (unsigned short*)carve((size_t)Ec * Hc * Dc * 2);   // 16MB
  unsigned short* W1b = (unsigned short*)carve((size_t)Ec * Hc * Hc * 2);   // 16MB
  unsigned short* W2t = (unsigned short*)carve((size_t)Dc * Ec * Hc * 2);   // 16MB
  unsigned short* h   = (unsigned short*)carve((size_t)Ec * Bc * Hc * 2);   // 128MB
  unsigned short* z   = (unsigned short*)carve((size_t)Ec * Bc * Hc * 2);   // 128MB
  unsigned short* h3  = (unsigned short*)carve((size_t)Bc * Ec * Hc * 2);   // 128MB
  float*          gate= (float*)carve((size_t)Bc * Ec * 4);                 // 256KB
  float*          gb  = (float*)carve((size_t)Bc * Dc * 4);                 // 32MB

  // precision conversions
  k_f2bf4<<<(Bc * Dc) / 1024, 256, 0, stream>>>(x, xb);
  k_f2bf4<<<(Ec * Hc * Dc) / 1024, 256, 0, stream>>>(W0, W0b);
  k_f2bf4<<<(Ec * Hc * Hc) / 1024, 256, 0, stream>>>(W1, W1b);
  k_w2t<<<(Ec * Dc * Hc) / 256, 256, 0, stream>>>(W2, W2t);

  // gating + combined output bias
  k_gate<<<Bc, 256, 0, stream>>>(x, Wg1, bg1, Wg2, bg2, gate);
  k_gbias<<<(Bc * Dc) / 256, 256, 0, stream>>>(gate, b2, gb);

  // layer 0: h[e] = relu(x W0[e]^T + b0[e])      [8192x1024, K=1024] x8
  moe_gemm<0><<<dim3(Hc / BN, Bc / BM, Ec), 256, 0, stream>>>(
      xb, W0b, b0, h, nullptr, Bc, Hc, Dc,
      0l, (long)Hc * Dc, (long)Hc, (long)Bc * Hc);

  // layer 1: z[e] = h[e] W1[e]^T + b1[e]         [8192x1024, K=1024] x8
  moe_gemm<1><<<dim3(Hc / BN, Bc / BM, Ec), 256, 0, stream>>>(
      h, W1b, b1, z, nullptr, Bc, Hc, Hc,
      (long)Bc * Hc, (long)Hc * Hc, (long)Hc, (long)Bc * Hc);

  // LN -> relu -> +h -> * gate  => h3[b, e*H+f]
  k_ln<<<dim3(Bc, Ec), 256, 0, stream>>>(z, h, lng, lnb, gate, h3);

  // combine: out = h3[B, E*H] W2t[D, E*H]^T + sum_e gate*b2   [K=8192]
  moe_gemm<2><<<dim3(Dc / BN, Bc / BM, 1), 256, 0, stream>>>(
      h3, W2t, gb, nullptr, out, Bc, Dc, Ec * Hc, 0l, 0l, 0l, 0l);
}
